// LPVModel5_29970281791603
// MI455X (gfx1250) — compile-verified, matrix operand order checked by reading, not run on hardware
//
#include <hip/hip_runtime.h>
#include <hip/hip_bf16.h>

// ---- problem constants (from reference) ----
#define BB    512
#define HH    50
#define TT    1024
#define STATE 13
#define CTRL  4
#define LAT   128
#define HID   256
#define Q0    3

typedef __attribute__((ext_vector_type(16))) __bf16          bf16x16;
typedef __attribute__((ext_vector_type(16))) unsigned short  u16x16;
typedef __attribute__((ext_vector_type(8)))  float           f32x8;

// round-to-nearest-even f32 -> bf16 bits
__device__ __forceinline__ unsigned short f2bf_bits(float f) {
  unsigned u = __float_as_uint(f);
  unsigned r = u + 0x7FFFu + ((u >> 16) & 1u);
  return (unsigned short)(r >> 16);
}
__device__ __forceinline__ float bf_bits2f(unsigned short b) {
  return __uint_as_float(((unsigned)b) << 16);
}

// ---------------------------------------------------------------------------
// Kernel 1: per-batch context mean -> h = tanh(m @ W_ctxt + b_ctxt)
// Also emits h split into bf16 hi/lo, pre-swizzled into the CDNA5 WMMA
// A-fragment layout (16-bit A 16x32, wave32): lane L holds M=L%16,
// element e holds K = e + (e>=8?8:0) + (L>=16?8:0).
// ---------------------------------------------------------------------------
__global__ __launch_bounds__(HID)
void ctx_h_kernel(const float* __restrict__ x_hist, const float* __restrict__ u_hist,
                  const float* __restrict__ W_ctxt, const float* __restrict__ b_ctxt,
                  float* __restrict__ h_out,
                  unsigned short* __restrict__ hsw_hi, unsigned short* __restrict__ hsw_lo) {
  int b = blockIdx.x;
  int t = threadIdx.x;
  __shared__ float m[STATE + CTRL];
  if (t < STATE + CTRL) {
    float s = 0.f;
    if (t < STATE) {
      const float* p = x_hist + (size_t)b * HH * STATE + t;
      for (int i = 0; i < HH; ++i) s += p[i * STATE];
    } else {
      const float* p = u_hist + (size_t)b * HH * CTRL + (t - STATE);
      for (int i = 0; i < HH; ++i) s += p[i * CTRL];
    }
    m[t] = s * (1.0f / (float)HH);
  }
  __syncthreads();
  float acc = b_ctxt[t];
#pragma unroll
  for (int k = 0; k < STATE + CTRL; ++k) acc += m[k] * W_ctxt[k * HID + t];
  float h = tanhf(acc);
  h_out[b * HID + t] = h;

  // swizzle destination for (batch=b, K=t)
  int kt = t >> 5, kk = t & 31;
  int lanehi = (kk & 8) ? 1 : 0;
  int e  = (kk & 7) + ((kk >= 16) ? 8 : 0);
  int bt = b >> 4;
  int L  = (b & 15) + 16 * lanehi;
  int di = ((bt * 8 + kt) * 32 + L) * 16 + e;

  unsigned short hb = f2bf_bits(h);
  float lo = h - bf_bits2f(hb);
  hsw_hi[di] = hb;
  hsw_lo[di] = f2bf_bits(lo);
}

// ---------------------------------------------------------------------------
// Kernel 1b: split W_A (256 x 16384) into bf16 hi/lo, pre-swizzled into the
// CDNA5 WMMA B-fragment layout (16-bit B 32x16, wave32): lane L holds
// N=L%16, element e holds K = e + (e>=8?8:0) + (L>=16?8:0).
// ---------------------------------------------------------------------------
__global__ __launch_bounds__(256)
void wa_swizzle_kernel(const float* __restrict__ W_A,
                       unsigned short* __restrict__ hi, unsigned short* __restrict__ lo) {
  int idx = blockIdx.x * 256 + threadIdx.x;     // 0 .. 4194303
  int e  = idx & 15;
  int L  = (idx >> 4) & 31;
  int kt = (idx >> 9) & 7;
  int nt = idx >> 12;
  int kk = e + ((e >= 8) ? 8 : 0) + ((L >= 16) ? 8 : 0);
  int K  = kt * 32 + kk;
  int N  = nt * 16 + (L & 15);
  float v = W_A[(size_t)K * (LAT * LAT) + N];
  unsigned short hb = f2bf_bits(v);
  hi[idx] = hb;
  lo[idx] = f2bf_bits(v - bf_bits2f(hb));
}

// ---------------------------------------------------------------------------
// Kernel 2: A_dt (512 x 16384) = I + h @ W_A + b_A via bf16x3 WMMA.
// Each wave owns a 1x4 strip of 16x16 tiles (1 batch-tile x 4 N-tiles):
// the h A-fragments are loaded once per K-step and reused for 4 N-tiles,
// so each K-step is 10 x 32B loads feeding 12 v_wmma_f32_16x16x32_bf16.
// bf16x3 split (hi*hi + hi*lo + lo*hi) recovers ~fp32 accuracy, which
// matters because errors in A compound through the 1024-step scan.
// ---------------------------------------------------------------------------
__global__ __launch_bounds__(256)
void gen_A_kernel(const unsigned short* __restrict__ hsw_hi, const unsigned short* __restrict__ hsw_lo,
                  const unsigned short* __restrict__ wsw_hi, const unsigned short* __restrict__ wsw_lo,
                  const float* __restrict__ b_A, float* __restrict__ A_out) {
  int wave = threadIdx.x >> 5;
  int lane = threadIdx.x & 31;
  int tg   = blockIdx.x * 8 + wave;   // 0 .. 8191 tile-groups
  int bt   = tg >> 8;                 // batch tile 0..31
  int ntb  = (tg & 255) * 4;          // first of 4 N tiles

  const unsigned short* ah = hsw_hi + ((size_t)(bt * 8) * 32 + lane) * 16;
  const unsigned short* al = hsw_lo + ((size_t)(bt * 8) * 32 + lane) * 16;
  const unsigned short* bh = wsw_hi + ((size_t)(ntb * 8) * 32 + lane) * 16;
  const unsigned short* bl = wsw_lo + ((size_t)(ntb * 8) * 32 + lane) * 16;

  f32x8 acc[4];
#pragma unroll
  for (int u = 0; u < 4; ++u) acc[u] = (f32x8){0.f, 0.f, 0.f, 0.f, 0.f, 0.f, 0.f, 0.f};

#pragma unroll
  for (int kt = 0; kt < 8; ++kt) {
    bf16x16 aH = __builtin_bit_cast(bf16x16, *(const u16x16*)(ah + kt * 512));
    bf16x16 aL = __builtin_bit_cast(bf16x16, *(const u16x16*)(al + kt * 512));
    bf16x16 bH[4], bL[4];
#pragma unroll
    for (int u = 0; u < 4; ++u) {
      bH[u] = __builtin_bit_cast(bf16x16, *(const u16x16*)(bh + u * 4096 + kt * 512));
      bL[u] = __builtin_bit_cast(bf16x16, *(const u16x16*)(bl + u * 4096 + kt * 512));
    }
#pragma unroll
    for (int u = 0; u < 4; ++u) {
      acc[u] = __builtin_amdgcn_wmma_f32_16x16x32_bf16(false, aH, false, bH[u], (short)0, acc[u], false, false);
      acc[u] = __builtin_amdgcn_wmma_f32_16x16x32_bf16(false, aH, false, bL[u], (short)0, acc[u], false, false);
      acc[u] = __builtin_amdgcn_wmma_f32_16x16x32_bf16(false, aL, false, bH[u], (short)0, acc[u], false, false);
    }
  }

  // C layout: lanes 0-15 -> M = v, lanes 16-31 -> M = 8+v; N = lane%16
  int N = lane & 15;
  int batch0 = bt * 16 + ((lane >= 16) ? 8 : 0);
#pragma unroll
  for (int u = 0; u < 4; ++u) {
    int n = (ntb + u) * 16 + N;
    float bias = b_A[n] + (((n % (LAT + 1)) == 0) ? 1.0f : 0.0f);  // + b_A + I
#pragma unroll
    for (int v = 0; v < 8; ++v) {
      A_out[(size_t)(batch0 + v) * (LAT * LAT) + n] = acc[u][v] + bias;
    }
  }
}

// ---------------------------------------------------------------------------
// Kernel 3: per-batch scan + fused decode.
// 256 threads (8 waves): thread t -> row r = t&127, K-half = t>>7.
// A row-half lives in VGPRs (64 floats); z ping-pongs in LDS (broadcast
// reads: waves 0-3 read half 0, waves 4-7 read half 1 -> same address per
// wave, conflict-free). While half 0 finalizes z_{k+1}, half 1 decodes z_k.
// ---------------------------------------------------------------------------
__global__ __launch_bounds__(256)
void scan_kernel(const float* __restrict__ A_dt, const float* __restrict__ h_ws,
                 const float* __restrict__ W_B, const float* __restrict__ b_B,
                 const float* __restrict__ x_init, const float* __restrict__ W_enc,
                 const float* __restrict__ b_enc, const float* __restrict__ u_future,
                 const float* __restrict__ W_dec, const float* __restrict__ b_dec,
                 float* __restrict__ out) {
  __shared__ __attribute__((aligned(16))) float u_lds[TT * CTRL];   // 16 KB
  __shared__ __attribute__((aligned(16))) float zbuf[2][LAT];
  __shared__ float part[LAT];
  __shared__ float wdecT[STATE * LAT];                              // [out][k]
  __shared__ float h_l[HID];
  __shared__ float bm[LAT * CTRL];
  __shared__ float dec_part[4][STATE];

  int b = blockIdx.x;
  int t = threadIdx.x;
  int r = t & (LAT - 1);
  int half = t >> 7;

  // warm the cache for this block's A rows while we do the setup math
  const float* arow_g = A_dt + (size_t)b * (LAT * LAT) + r * LAT + half * 64;
  __builtin_prefetch(arow_g, 0, 3);

  h_l[t] = h_ws[b * HID + t];
  {
    const float4* usrc = (const float4*)(u_future + (size_t)b * TT * CTRL);
    float4* udst = (float4*)u_lds;
    for (int i = t; i < TT; i += 256) udst[i] = usrc[i];
  }
  for (int i = t; i < STATE * LAT; i += 256) {
    int oi = i >> 7, ci = i & (LAT - 1);
    wdecT[i] = W_dec[ci * STATE + oi];
  }
  __syncthreads();

  // B_mat = h @ W_B + b_B  (128x4 per batch)
  for (int o = t; o < LAT * CTRL; o += 256) {
    float s = b_B[o];
    for (int k = 0; k < HID; ++k) s += h_l[k] * W_B[k * (LAT * CTRL) + o];
    bm[o] = s;
  }
  // z0 = x_init @ W_enc + b_enc
  if (t < LAT) {
    float s = b_enc[t];
    const float* xi = x_init + b * STATE;
#pragma unroll
    for (int s2 = 0; s2 < STATE; ++s2) s += xi[s2] * W_enc[s2 * LAT + t];
    zbuf[0][t] = s;
  }
  __syncthreads();

  // A row-half -> registers (16 x float4 = 64 floats)
  float4 Arow[16];
  {
    const float4* ap = (const float4*)arow_g;
#pragma unroll
    for (int i = 0; i < 16; ++i) Arow[i] = ap[i];
  }
  float bm0 = 0.f, bm1 = 0.f, bm2 = 0.f, bm3 = 0.f;
  if (half == 0) { bm0 = bm[r*4+0]; bm1 = bm[r*4+1]; bm2 = bm[r*4+2]; bm3 = bm[r*4+3]; }

  int j    = t - 128;        // decode role for half==1
  int dout = j % STATE;
  int seg  = j / STATE;      // valid when 0 <= j < 4*STATE

  int cur = 0;
  for (int k = 0; k <= TT; ++k) {
    float acc = 0.f;
    if (k < TT) {
      const float4* zp = (const float4*)&zbuf[cur][half * 64];
#pragma unroll
      for (int i = 0; i < 16; ++i) {
        float4 a = Arow[i];
        float4 z = zp[i];
        acc += a.x * z.x + a.y * z.y + a.z * z.z + a.w * z.w;
      }
      if (half == 1) part[r] = acc;
    }
    __syncthreads();
    if (half == 0) {
      if (k < TT) {
        float u0 = u_lds[k*4+0], u1 = u_lds[k*4+1], u2 = u_lds[k*4+2], u3 = u_lds[k*4+3];
        zbuf[cur ^ 1][r] = acc + part[r] + bm0*u0 + bm1*u1 + bm2*u2 + bm3*u3;
      }
    } else if (j < 4 * STATE) {
      // decode z_k while the other half builds z_{k+1}
      float s = 0.f;
      const float* zc = &zbuf[cur][seg * 32];
      const float* wp = &wdecT[dout * LAT + seg * 32];
#pragma unroll
      for (int i = 0; i < 32; ++i) s += zc[i] * wp[i];
      dec_part[seg][dout] = s;
    }
    __syncthreads();
    if (j >= 0 && j < STATE) {
      float x = b_dec[j] + dec_part[0][j] + dec_part[1][j] + dec_part[2][j] + dec_part[3][j];
      out[(size_t)b * (TT + 1) * STATE + (size_t)k * STATE + j] = x;
    }
    cur ^= 1;
  }
}

// ---------------------------------------------------------------------------
// Kernel 4: in-place L2-normalize out[..., Q0:Q0+4]
// ---------------------------------------------------------------------------
__global__ __launch_bounds__(256)
void norm_kernel(float* __restrict__ out) {
  int row = blockIdx.x * 256 + threadIdx.x;
  if (row >= BB * (TT + 1)) return;
  float* p = out + (size_t)row * STATE + Q0;
  float a = p[0], b = p[1], c = p[2], d = p[3];
  float n = sqrtf(a*a + b*b + c*c + d*d);
  float inv = 1.0f / fmaxf(n, 1e-12f);
  p[0] = a * inv; p[1] = b * inv; p[2] = c * inv; p[3] = d * inv;
}

// ---------------------------------------------------------------------------
extern "C" void kernel_launch(void* const* d_in, const int* in_sizes, int n_in,
                              void* d_out, int out_size, void* d_ws, size_t ws_size,
                              hipStream_t stream) {
  (void)in_sizes; (void)n_in; (void)out_size; (void)ws_size;

  const float* x_history = (const float*)d_in[0];
  const float* u_history = (const float*)d_in[1];
  const float* x_init    = (const float*)d_in[2];
  /* d_in[3] = x_future: unused by the reference */
  const float* u_future  = (const float*)d_in[4];
  /* d_in[5] = n_steps (==1024): compile-time constant here */
  const float* W_ctxt = (const float*)d_in[6];
  const float* b_ctxt = (const float*)d_in[7];
  const float* W_A    = (const float*)d_in[8];
  const float* b_A    = (const float*)d_in[9];
  const float* W_B    = (const float*)d_in[10];
  const float* b_B    = (const float*)d_in[11];
  const float* W_enc  = (const float*)d_in[12];
  const float* b_enc  = (const float*)d_in[13];
  const float* W_dec  = (const float*)d_in[14];
  const float* b_dec  = (const float*)d_in[15];
  float* out = (float*)d_out;

  // workspace layout
  char* ws = (char*)d_ws;
  float*          h_ws   = (float*)(ws + 0);                 // 512*256*4        = 0.5 MB
  unsigned short* hsw_hi = (unsigned short*)(ws + 524288);   // 131072*2         = 256 KB
  unsigned short* hsw_lo = (unsigned short*)(ws + 786432);   // 256 KB
  unsigned short* wsw_hi = (unsigned short*)(ws + 1048576);  // 4194304*2        = 8 MB
  unsigned short* wsw_lo = (unsigned short*)(ws + 9437184);  // 8 MB
  float*          A_ws   = (float*)(ws + 17825792);          // 512*16384*4      = 33.5 MB

  ctx_h_kernel<<<BB, HID, 0, stream>>>(x_history, u_history, W_ctxt, b_ctxt,
                                       h_ws, hsw_hi, hsw_lo);
  wa_swizzle_kernel<<<16384, 256, 0, stream>>>(W_A, wsw_hi, wsw_lo);
  gen_A_kernel<<<1024, 256, 0, stream>>>(hsw_hi, hsw_lo, wsw_hi, wsw_lo, b_A, A_ws);
  scan_kernel<<<BB, 256, 0, stream>>>(A_ws, h_ws, W_B, b_B, x_init, W_enc, b_enc,
                                      u_future, W_dec, b_dec, out);
  norm_kernel<<<(BB * (TT + 1) + 255) / 256, 256, 0, stream>>>(out);
}
